// PaddedAttentionTT_70781061038505
// MI455X (gfx1250) — compile-verified, matrix operand order checked by reading, not run on hardware
//
#include <hip/hip_runtime.h>
#include <stdint.h>

// Problem constants (Falcon-7B-ish padded attention)
#define S_LEN 2048
#define HID   4608
#define NHEAD 72
#define HDIM  64

#define LDS_STRIDE 40   // 32 bf16 payload + 8 pad -> 80B row pitch, conflict-free b128 frag reads

#if defined(__HIP_DEVICE_COMPILE__) && __has_builtin(__builtin_amdgcn_global_load_async_to_lds_b128)
#define HAVE_ASYNC_LDS 1
#else
#define HAVE_ASYNC_LDS 0
#endif

typedef __attribute__((ext_vector_type(16))) __bf16 v16bf;
typedef __attribute__((ext_vector_type(8)))  float  v8f;
typedef __attribute__((ext_vector_type(4)))  int    v4i_t;
#if HAVE_ASYNC_LDS
typedef __attribute__((address_space(1))) v4i_t gv4i_t;   // global v4i
typedef __attribute__((address_space(3))) v4i_t lv4i_t;   // LDS v4i
#endif

// ---------- bf16 helpers (bit-exact RNE, no reliance on __bf16 arithmetic) ----------
__device__ __forceinline__ __bf16 f2bf(float x) {
    union { float f; uint32_t u; } a; a.f = x;
    uint32_t r = (a.u + 0x7FFFu + ((a.u >> 16) & 1u)) >> 16;
    union { unsigned short s; __bf16 b; } o; o.s = (unsigned short)r;
    return o.b;
}
__device__ __forceinline__ float bf2f(__bf16 b) {
    union { unsigned short s; __bf16 b; } i; i.b = b;
    union { uint32_t u; float f; } o; o.u = ((uint32_t)i.s) << 16;
    return o.f;
}

// ---------- WMMA fragment load (CDNA5 16-bit A/B layout) ----------
// Lane L (L<16): row M=L, holds K chunks [0..7] and [16..23]
// Lane L (L>=16): row M=L-16, holds K chunks [8..15] and [24..31]
union FragU { uint4 q[2]; v16bf v; };

__device__ __forceinline__ v16bf load_frag(const __bf16* base, int ld) {
    const int lane = threadIdx.x & 31;
    const int row  = lane & 15;
    const int hi   = lane >> 4;
    const __bf16* p = base + (size_t)row * ld + hi * 8;
    FragU f;
    f.q[0] = *reinterpret_cast<const uint4*>(p);
    f.q[1] = *reinterpret_cast<const uint4*>(p + 16);
    return f.v;
}

__device__ __forceinline__ v8f wmma_bf16(v16bf a, v16bf b, v8f c) {
    return __builtin_amdgcn_wmma_f32_16x16x32_bf16(false, a, false, b, (short)0, c, false, false);
}

__device__ __forceinline__ void store_elem(float* p, float v)  { *p = v; }
__device__ __forceinline__ void store_elem(__bf16* p, float v) { *p = f2bf(v); }

// ---------- async tile staging: rows x 32 bf16 (row-major, ld) -> LDS (pitch LDS_STRIDE) ----------
__device__ __forceinline__ void copy_tile_to_lds(const __bf16* __restrict__ g, int ld,
                                                 __bf16* l, int rows) {
    const int nchunk = rows * 4;                 // 16B chunks: 4 per 64B row
    for (int idx = threadIdx.x; idx < nchunk; idx += blockDim.x) {
        const int r = idx >> 2, c = idx & 3;
        const __bf16* gp = g + (size_t)r * ld + c * 8;
        __bf16* lp = l + r * LDS_STRIDE + c * 8;
#if HAVE_ASYNC_LDS
        __builtin_amdgcn_global_load_async_to_lds_b128(
            (gv4i_t*)(uintptr_t)gp, (lv4i_t*)lp, 0, 0);
#else
        *reinterpret_cast<uint4*>(lp) = *reinterpret_cast<const uint4*>(gp);
#endif
    }
}

__device__ __forceinline__ void wait_async_lds() {
#if HAVE_ASYNC_LDS
#if __has_builtin(__builtin_amdgcn_s_wait_asynccnt)
    __builtin_amdgcn_s_wait_asynccnt(0);
#else
    asm volatile("s_wait_asynccnt 0x0" ::: "memory");
#endif
#endif
}

// ---------- f32 -> bf16 conversion ----------
__global__ void cvt_f32_bf16(const float* __restrict__ src, __bf16* __restrict__ dst, int n) {
    int i = blockIdx.x * blockDim.x + threadIdx.x;
    if (i < n) dst[i] = f2bf(src[i]);
}

// ---------- GEMM: C[M,N] = A[M,K] @ B[N,K]^T  (bf16 in, f32 accumulate) ----------
// Block = 4 waves. Block tile 64(M) x 256(N); each wave owns 64x64 (16 WMMA accumulators).
// A(64x32) and B(256x32) k-slabs are double-buffered in LDS, filled with async-to-LDS copies.
template <typename OutT>
__global__ __launch_bounds__(128) void gemm_bt_wmma(
    const __bf16* __restrict__ A, const __bf16* __restrict__ B,
    OutT* __restrict__ C, int M, int N, int K) {
    __shared__ __bf16 aT[2][64 * LDS_STRIDE];
    __shared__ __bf16 bT[2][256 * LDS_STRIDE];

    const int wave = threadIdx.x >> 5;
    const int lane = threadIdx.x & 31;
    const int m0     = blockIdx.x * 64;
    const int nBlock = blockIdx.y * 256;
    const int n0     = nBlock + wave * 64;
    const bool active = (m0 < M) && (n0 < N);
    int bRows = N - nBlock; if (bRows > 256) bRows = 256;

    // prologue fill of buffer 0
    copy_tile_to_lds(A + (size_t)m0 * K, K, aT[0], 64);
    copy_tile_to_lds(B + (size_t)nBlock * K, K, bT[0], bRows);

    v8f acc[4][4] = {};
    const int steps = K / 32;
    for (int j = 0; j < steps; ++j) {
        const int cur = j & 1;
        wait_async_lds();
        __syncthreads();                       // buffer `cur` ready; prev compute on cur^1 done
        if (j + 1 < steps) {                   // kick off next slab into the other buffer
            const int k0 = (j + 1) * 32;
            copy_tile_to_lds(A + (size_t)m0 * K + k0, K, aT[cur ^ 1], 64);
            copy_tile_to_lds(B + (size_t)nBlock * K + k0, K, bT[cur ^ 1], bRows);
        }
        if (active) {
            v16bf af[4];
#pragma unroll
            for (int i = 0; i < 4; ++i)
                af[i] = load_frag(aT[cur] + i * 16 * LDS_STRIDE, LDS_STRIDE);
#pragma unroll
            for (int t = 0; t < 4; ++t) {
                v16bf bf = load_frag(bT[cur] + (wave * 64 + t * 16) * LDS_STRIDE, LDS_STRIDE);
#pragma unroll
                for (int i = 0; i < 4; ++i)
                    acc[i][t] = wmma_bf16(af[i], bf, acc[i][t]);
            }
        }
        __syncthreads();                       // all waves done reading `cur` before overwrite
    }

    if (active) {
        // C/D layout: VGPR r, lane L -> M = r + 8*(L>=16), N = L&15
        const int n  = lane & 15;
        const int mb = (lane >> 4) * 8;
#pragma unroll
        for (int i = 0; i < 4; ++i)
#pragma unroll
            for (int t = 0; t < 4; ++t)
#pragma unroll
                for (int r = 0; r < 8; ++r)
                    store_elem(&C[(size_t)(m0 + i * 16 + mb + r) * N + n0 + t * 16 + n],
                               acc[i][t][r]);
    }
}

// ---------- RoPE on Q (bf16 in-place). rotate_half: (x1,x2)->(-x2,x1), split at 32 ----------
__global__ void rope_q_kernel(__bf16* __restrict__ q,
                              const float* __restrict__ cosb, const float* __restrict__ sinb) {
    int i = blockIdx.x * blockDim.x + threadIdx.x;
    if (i >= S_LEN * NHEAD * 32) return;
    int d = i & 31;
    int h = (i >> 5) % NHEAD;
    int s = i / (32 * NHEAD);
    size_t base = (size_t)s * HID + h * HDIM;
    float q0 = bf2f(q[base + d]);
    float q1 = bf2f(q[base + d + 32]);
    float c0 = cosb[s * HDIM + d], c1 = cosb[s * HDIM + d + 32];
    float s0 = sinb[s * HDIM + d], s1 = sinb[s * HDIM + d + 32];
    q[base + d]      = f2bf(q0 * c0 - q1 * s0);
    q[base + d + 32] = f2bf(q1 * c1 + q0 * s1);
}

// ---------- RoPE on K + emit key_ret/val_ret (f32 outputs) + bf16 K + bf16 V^T ----------
__global__ void rope_kv_kernel(const float* __restrict__ kf, const float* __restrict__ vf,
                               const float* __restrict__ cosb, const float* __restrict__ sinb,
                               float* __restrict__ key_ret, float* __restrict__ val_ret,
                               __bf16* __restrict__ kb, __bf16* __restrict__ vtb) {
    int i = blockIdx.x * blockDim.x + threadIdx.x;
    if (i >= S_LEN * 32) return;
    int d = i & 31;
    int s = i >> 5;
    float c0 = cosb[s * HDIM + d], c1 = cosb[s * HDIM + d + 32];
    float s0 = sinb[s * HDIM + d], s1 = sinb[s * HDIM + d + 32];
    float k0 = kf[s * HDIM + d], k1 = kf[s * HDIM + d + 32];
    float nk0 = k0 * c0 - k1 * s0;
    float nk1 = k1 * c1 + k0 * s1;
    key_ret[s * HDIM + d]      = nk0;
    key_ret[s * HDIM + d + 32] = nk1;
    kb[s * HDIM + d]      = f2bf(nk0);
    kb[s * HDIM + d + 32] = f2bf(nk1);
    float v0 = vf[s * HDIM + d], v1 = vf[s * HDIM + d + 32];
    val_ret[s * HDIM + d]      = v0;
    val_ret[s * HDIM + d + 32] = v1;
    vtb[(size_t)d * S_LEN + s]        = f2bf(v0);   // V^T [64][S] -> contiguous B-frags for P@V
    vtb[(size_t)(d + 32) * S_LEN + s] = f2bf(v1);
}

// ---------- Flash-style causal multi-query attention ----------
// 4 waves / block, each wave owns one 16-row q-tile of the same head (K/V stream shared in L1).
// Per 32-key step: 4 WMMA scores + online softmax + LDS C->A transpose + 4 WMMA PV.
// P staging is per-wave LDS; same-wave LDS ops are in-order, so no block barrier needed.
__global__ __launch_bounds__(128) void attn_kernel(
    const __bf16* __restrict__ qb, const __bf16* __restrict__ kb,
    const __bf16* __restrict__ vtb, __bf16* __restrict__ attn) {
    const int wave = threadIdx.x >> 5;
    const int qi = blockIdx.x * 4 + wave;
    const int h  = blockIdx.y;
    const int m0 = qi * 16;
    const int lane = threadIdx.x & 31;
    const int n  = lane & 15;
    const int hi = lane >> 4;

    __shared__ __bf16 pbuf[4][16 * 32];   // per-wave P tile, C-layout -> A-layout staging
    __bf16* pw = pbuf[wave];

    // Q tile fragments (K=64 split into two 32-deep chunks)
    v16bf qa0 = load_frag(qb + (size_t)m0 * HID + h * HDIM,      HID);
    v16bf qa1 = load_frag(qb + (size_t)m0 * HID + h * HDIM + 32, HID);

    v8f o[4] = {};
    float mrow[8], lrow[8];
#pragma unroll
    for (int r = 0; r < 8; ++r) { mrow[r] = -1e30f; lrow[r] = 0.0f; }

    const int nj = (m0 + 47) / 32;   // causal: keys [0, m0+15]
    for (int j = 0; j < nj; ++j) {
        const int kbase = j * 32;
        // scores: two 16x16 n-tiles, each accumulated over d=64 (2 WMMA each)
        v8f s0v = {}, s1v = {};
        {
            v16bf b00 = load_frag(kb + (size_t)kbase * HDIM,      HDIM);
            v16bf b01 = load_frag(kb + (size_t)kbase * HDIM + 32, HDIM);
            s0v = wmma_bf16(qa0, b00, s0v);
            s0v = wmma_bf16(qa1, b01, s0v);
            v16bf b10 = load_frag(kb + (size_t)(kbase + 16) * HDIM,      HDIM);
            v16bf b11 = load_frag(kb + (size_t)(kbase + 16) * HDIM + 32, HDIM);
            s1v = wmma_bf16(qa0, b10, s1v);
            s1v = wmma_bf16(qa1, b11, s1v);
        }
        const float scale = 0.125f;   // 1/sqrt(64)
#pragma unroll
        for (int r = 0; r < 8; ++r) {
            const int qrow = m0 + hi * 8 + r;
            float a = s0v[r] * scale;
            float b = s1v[r] * scale;
            if (kbase + n      > qrow) a = -1e30f;
            if (kbase + 16 + n > qrow) b = -1e30f;
            // row max across the 16 lanes of this half-wave
            float mx = fmaxf(a, b);
            mx = fmaxf(mx, __shfl_xor(mx, 1, 32));
            mx = fmaxf(mx, __shfl_xor(mx, 2, 32));
            mx = fmaxf(mx, __shfl_xor(mx, 4, 32));
            mx = fmaxf(mx, __shfl_xor(mx, 8, 32));
            float newm = fmaxf(mrow[r], mx);
            float p0 = __expf(a - newm);
            float p1 = __expf(b - newm);
            float ps = p0 + p1;
            ps += __shfl_xor(ps, 1, 32);
            ps += __shfl_xor(ps, 2, 32);
            ps += __shfl_xor(ps, 4, 32);
            ps += __shfl_xor(ps, 8, 32);
            float corr = __expf(mrow[r] - newm);
            lrow[r] = lrow[r] * corr + ps;
            mrow[r] = newm;
            o[0][r] *= corr; o[1][r] *= corr; o[2][r] *= corr; o[3][r] *= corr;
            pw[(hi * 8 + r) * 32 + n]      = f2bf(p0);
            pw[(hi * 8 + r) * 32 + 16 + n] = f2bf(p1);
        }
        // P @ V : A = P [16x32] from per-wave LDS (same-wave LDS in-order), B = V^T tiles
        v16bf pa = load_frag(pw, 32);
#pragma unroll
        for (int t = 0; t < 4; ++t) {
            v16bf vb = load_frag(vtb + (size_t)(t * 16) * S_LEN + kbase, S_LEN);
            o[t] = wmma_bf16(pa, vb, o[t]);
        }
    }
    // epilogue: normalize and write attn output tile (bf16, [S][72*64])
#pragma unroll
    for (int r = 0; r < 8; ++r) {
        float inv = 1.0f / lrow[r];
        size_t row = (size_t)(m0 + hi * 8 + r) * HID + h * HDIM;
#pragma unroll
        for (int t = 0; t < 4; ++t)
            attn[row + t * 16 + n] = f2bf(o[t][r] * inv);
    }
}

extern "C" void kernel_launch(void* const* d_in, const int* in_sizes, int n_in,
                              void* d_out, int out_size, void* d_ws, size_t ws_size,
                              hipStream_t stream) {
    const float* hs   = (const float*)d_in[0];
    const float* cosb = (const float*)d_in[1];
    const float* sinb = (const float*)d_in[2];
    const float* wq   = (const float*)d_in[3];
    const float* wk   = (const float*)d_in[4];
    const float* wv   = (const float*)d_in[5];
    const float* wd   = (const float*)d_in[6];

    float* out     = (float*)d_out;                       // [S, HID]
    float* key_ret = out + (size_t)S_LEN * HID;           // [S, 64]
    float* val_ret = key_ret + (size_t)S_LEN * HDIM;      // [S, 64]

    char* ws = (char*)d_ws;
    size_t off = 0;
    auto alloc = [&](size_t bytes) {
        char* p = ws + off;
        off = (off + bytes + 255) & ~(size_t)255;
        return p;
    };
    __bf16* hsb  = (__bf16*)alloc((size_t)S_LEN * HID * 2);
    __bf16* wqb  = (__bf16*)alloc((size_t)HID * HID * 2);
    __bf16* wkb  = (__bf16*)alloc((size_t)HDIM * HID * 2);
    __bf16* wvb  = (__bf16*)alloc((size_t)HDIM * HID * 2);
    __bf16* wdb  = (__bf16*)alloc((size_t)HID * HID * 2);
    __bf16* qb   = (__bf16*)alloc((size_t)S_LEN * HID * 2);
    float*  kf   = (float*) alloc((size_t)S_LEN * HDIM * 4);
    float*  vf   = (float*) alloc((size_t)S_LEN * HDIM * 4);
    __bf16* kb   = (__bf16*)alloc((size_t)S_LEN * HDIM * 2);
    __bf16* vtb  = (__bf16*)alloc((size_t)HDIM * S_LEN * 2);
    __bf16* attn = (__bf16*)alloc((size_t)S_LEN * HID * 2);
    (void)ws_size; (void)in_sizes; (void)n_in; (void)out_size;

    auto cvt = [&](const float* s, __bf16* d, size_t n) {
        cvt_f32_bf16<<<dim3((unsigned)((n + 255) / 256)), dim3(256), 0, stream>>>(s, d, (int)n);
    };
    cvt(hs, hsb, (size_t)S_LEN * HID);
    cvt(wq, wqb, (size_t)HID * HID);
    cvt(wk, wkb, (size_t)HDIM * HID);
    cvt(wv, wvb, (size_t)HDIM * HID);
    cvt(wd, wdb, (size_t)HID * HID);

    // Q = hs @ wq^T  -> bf16 (RoPE applied in-place next)
    gemm_bt_wmma<__bf16><<<dim3(S_LEN / 64, HID / 256), 128, 0, stream>>>(hsb, wqb, qb, S_LEN, HID, HID);
    // K,V = hs @ w{k,v}^T -> f32 (needed for f32 key/val outputs)
    gemm_bt_wmma<float><<<dim3(S_LEN / 64, 1), 128, 0, stream>>>(hsb, wkb, kf, S_LEN, HDIM, HID);
    gemm_bt_wmma<float><<<dim3(S_LEN / 64, 1), 128, 0, stream>>>(hsb, wvb, vf, S_LEN, HDIM, HID);

    rope_q_kernel<<<(S_LEN * NHEAD * 32 + 255) / 256, 256, 0, stream>>>(qb, cosb, sinb);
    rope_kv_kernel<<<(S_LEN * 32 + 255) / 256, 256, 0, stream>>>(kf, vf, cosb, sinb,
                                                                 key_ret, val_ret, kb, vtb);

    // causal multi-query attention: (q-tile groups, heads)
    attn_kernel<<<dim3(S_LEN / 64, NHEAD), 128, 0, stream>>>(qb, kb, vtb, attn);

    // out = attn @ wd^T -> f32 into d_out
    gemm_bt_wmma<float><<<dim3(S_LEN / 64, HID / 256), 128, 0, stream>>>(attn, wdb, out, S_LEN, HID, HID);
}
